// TokenFNOBranch_30494267802195
// MI455X (gfx1250) — matrix operand order repaired and checked by reading, not compiled, not run on hardware
//
#include <hip/hip_runtime.h>
#include <math.h>

typedef __attribute__((ext_vector_type(16))) __bf16 v16bf;
typedef __attribute__((ext_vector_type(8)))  __bf16 v8bf;
typedef __attribute__((ext_vector_type(8)))  float  v8f;

#define DEV __device__ __forceinline__

static constexpr int Bb = 16, Hh = 128, Ww = 128, Cc = 64;
static constexpr int HW = Hh * Ww;        // 16384
static constexpr int NIMG = Bb * Cc;      // 1024
static constexpr int NTOK = Bb * HW;      // 262144
static constexpr float EPS = 1e-5f;
static constexpr float TWO_PI = 6.283185307179586f;

// ---------------- workspace layout (bytes) ----------------
static constexpr size_t MB = 1u << 20;
static constexpr size_t OFF_WAT = 0;            // bf16 [32][128]   fwd-W basis (B layout [N][K])
static constexpr size_t OFF_FBH = 8 * 1024;     // bf16 [3][32][128] fwd-H basis cos/sin/-sin (A layout)
static constexpr size_t OFF_FIH = 32 * 1024;    // bf16 [3][128][32] inv-H basis (A layout)
static constexpr size_t OFF_WET = 56 * 1024;    // bf16 [128][32]   inv-W basis (B layout [N][K])
static constexpr size_t OFF_W1B = 64 * 1024;    // bf16 [128][64]   mlp w1 (B layout)
static constexpr size_t OFF_W2B = 80 * 1024;    // bf16 [64][128]   mlp w2 (B layout)
static constexpr size_t OFF_ST1 = 96 * 1024;    // f32 [1024][2] mean,rstd
static constexpr size_t OFF_ST2 = 104 * 1024;   // f32 [1024][2]
static constexpr size_t OFF_P1  = 128 * 1024;   // f32 [16][64][64][2] partial sums (512KB)
static constexpr size_t OFF_XN  = 1 * MB;       // bf16 [img][16384]  normalized x   (32MB)
static constexpr size_t OFF_GT  = 33 * MB;      // bf16 [img][32][128] G transposed  (8MB)
static constexpr size_t OFF_AR  = 41 * MB;      // bf16 [512][16][64] X real         (1MB)
static constexpr size_t OFF_AI  = 42 * MB;      // bf16 [512][16][64] X imag         (1MB)
static constexpr size_t OFF_WR  = 43 * MB;      // bf16 [512][64][64] W real         (4MB)
static constexpr size_t OFF_WI  = 47 * MB;      // bf16 Wi                           (4MB)
static constexpr size_t OFF_WN  = 51 * MB;      // bf16 -Wi                          (4MB)
static constexpr size_t OFF_YTR = 55 * MB;      // bf16 [1024][16][32] Y real trans  (1MB)
static constexpr size_t OFF_YTI = 56 * MB;      // bf16 Y imag                       (1MB)
static constexpr size_t OFF_Z   = 57 * MB;      // bf16 [131072][32] Z rows          (8MB)
static constexpr size_t OFF_YS  = 65 * MB;      // bf16 [img2][16384] spectral out   (32MB)
static constexpr size_t OFF_YMLP = OFF_XN;      // reuse: bf16 [tok][64]             (32MB)
static constexpr size_t OFF_HACT = 33 * MB;     // reuse: bf16 [tok][128]            (64MB)
// peak = 97MB

// ---------------- WMMA fragment helpers ----------------
// A-matrix 16x32 bf16, row-major tile (ld elements per row).
// lane<16: row=lane,   K 0..15 ;  lane>=16: row=lane-16, K 16..31
DEV v16bf load_a(const __bf16* p, int ld) {
  int lane = threadIdx.x & 31;
  const __bf16* q = p + (size_t)(lane & 15) * ld + ((lane & 16) ? 16 : 0);
  v8bf lo = *(const v8bf*)(q);
  v8bf hi = *(const v8bf*)(q + 8);
  return __builtin_shufflevector(lo, hi, 0, 1, 2, 3, 4, 5, 6, 7, 8, 9, 10, 11, 12, 13, 14, 15);
}
// B-matrix 32x16 bf16 stored as [N][K] (column of B contiguous in K), ld = K stride.
// lane<16: col=lane, K 0..15 ;  lane>=16: col=lane-16, K 16..31
DEV v16bf load_b(const __bf16* p, int ld) {
  int lane = threadIdx.x & 31;
  const __bf16* q = p + (size_t)(lane & 15) * ld + ((lane & 16) ? 16 : 0);
  v8bf lo = *(const v8bf*)(q);
  v8bf hi = *(const v8bf*)(q + 8);
  return __builtin_shufflevector(lo, hi, 0, 1, 2, 3, 4, 5, 6, 7, 8, 9, 10, 11, 12, 13, 14, 15);
}
DEV v8f wmma_bf16(v16bf a, v16bf b, v8f c) {
  return __builtin_amdgcn_wmma_f32_16x16x32_bf16(false, a, false, b, (short)0, c, false, false);
}

// ---------------- table init ----------------
__global__ __launch_bounds__(256) void k_init_tables(char* ws, const float* mlp_w1,
                                                     const float* mlp_w2) {
  int i = blockIdx.x * 256 + threadIdx.x;  // 0..49151
  const float inv = 1.0f / 16384.0f;
  if (i < 4096) {  // WAT [32 n][128 w]: n<16 -> cos, n>=16 -> -sin  (e^{-i})
    int n = i >> 7, w = i & 127;
    int kx = (n < 16) ? n : n - 16;
    float ang = TWO_PI * (float)(kx * w) / 128.0f;
    float v = (n < 16) ? cosf(ang) : -sinf(ang);
    ((__bf16*)(ws + OFF_WAT))[i] = (__bf16)v;
  } else if (i < 16384) {  // FBH [3][32 j][128 h]
    int t = i - 4096;
    int which = t >> 12, r = t & 4095;
    int j = r >> 7, h = r & 127;
    int ky = (j < 16) ? j : 96 + j;  // 112..127
    float ang = TWO_PI * (float)(ky * h) / 128.0f;
    float v = (which == 0) ? cosf(ang) : ((which == 1) ? sinf(ang) : -sinf(ang));
    ((__bf16*)(ws + OFF_FBH))[t] = (__bf16)v;
  } else if (i < 28672) {  // FIH [3][128 h][32 j]
    int t = i - 16384;
    int which = t >> 12, r = t & 4095;
    int h = r >> 5, j = r & 31;
    int ky = (j < 16) ? j : 96 + j;
    float ang = TWO_PI * (float)(ky * h) / 128.0f;
    float v = (which == 0) ? cosf(ang) : ((which == 1) ? sinf(ang) : -sinf(ang));
    ((__bf16*)(ws + OFF_FIH))[t] = (__bf16)v;
  } else if (i < 32768) {  // WET [128 w][32 k]: k<16 c*cos, k>=16 -c*sin, scaled 1/HW
    int t = i - 28672;
    int w = t >> 5, k = t & 31;
    int kx = (k < 16) ? k : k - 16;
    float c = (kx == 0) ? 1.0f : 2.0f;
    float ang = TWO_PI * (float)(kx * w) / 128.0f;
    float v = (k < 16) ? (c * cosf(ang) * inv) : (-c * sinf(ang) * inv);
    ((__bf16*)(ws + OFF_WET))[t] = (__bf16)v;
  } else if (i < 40960) {  // W1B [128 o][64 c]
    int t = i - 32768;
    ((__bf16*)(ws + OFF_W1B))[t] = (__bf16)mlp_w1[t];
  } else if (i < 49152) {  // W2B [64 o][128 c]
    int t = i - 40960;
    ((__bf16*)(ws + OFF_W2B))[t] = (__bf16)mlp_w2[t];
  }
}

// ---------------- spectral weight pack: [ci][co][16][16] f32 -> [mode][co][ci] bf16 ----------------
__global__ __launch_bounds__(256) void k_packw(char* ws, const float* w1r, const float* w1i,
                                               const float* w2r, const float* w2i) {
  size_t i = (size_t)blockIdx.x * 256 + threadIdx.x;  // 0..2097151
  int mode = (int)(i >> 12);
  int rem = (int)(i & 4095);
  int co = rem >> 6, ci = rem & 63;
  int ky = mode >> 4, kx = mode & 15;
  int x = (ky < 16) ? ky : ky - 16;
  size_t src = (((size_t)(ci * 64 + co)) * 16 + x) * 16 + kx;
  float r = (ky < 16) ? w1r[src] : w2r[src];
  float im = (ky < 16) ? w1i[src] : w2i[src];
  ((__bf16*)(ws + OFF_WR))[i] = (__bf16)r;
  ((__bf16*)(ws + OFF_WI))[i] = (__bf16)im;
  ((__bf16*)(ws + OFF_WN))[i] = (__bf16)(-im);
}

// ---------------- instance norm 1: partial sums over token-major x ----------------
__global__ __launch_bounds__(256) void k_in1_partial(const float* x, char* ws) {
  __shared__ float ls[256], ls2[256];
  int b = blockIdx.x >> 6, chunk = blockIdx.x & 63;
  int t = threadIdx.x, c = t & 63, sub = t >> 6;
  int hwbase = chunk * 256 + sub * 64;
  float s = 0.f, s2 = 0.f;
  for (int i = 0; i < 64; i++) {
    float v = x[((size_t)b * HW + hwbase + i) * 64 + c];
    s += v;
    s2 += v * v;
  }
  ls[t] = s; ls2[t] = s2;
  __syncthreads();
  if (t < 64) {
    float S = 0.f, S2 = 0.f;
    for (int g = 0; g < 4; g++) { S += ls[g * 64 + t]; S2 += ls2[g * 64 + t]; }
    float* p1 = (float*)(ws + OFF_P1);
    size_t o = (((size_t)b * 64 + chunk) * 64 + t) * 2;
    p1[o] = S; p1[o + 1] = S2;
  }
}
__global__ __launch_bounds__(256) void k_in1_final(char* ws) {
  int i = blockIdx.x * 256 + threadIdx.x;  // 0..1023 = b*64+c
  if (i >= NIMG) return;
  int b = i >> 6, c = i & 63;
  const float* p1 = (const float*)(ws + OFF_P1);
  float S = 0.f, S2 = 0.f;
  for (int chunk = 0; chunk < 64; chunk++) {
    size_t o = (((size_t)b * 64 + chunk) * 64 + c) * 2;
    S += p1[o]; S2 += p1[o + 1];
  }
  float m = S / 16384.0f;
  float var = S2 / 16384.0f - m * m;
  float* st = (float*)(ws + OFF_ST1);
  st[i * 2] = m;
  st[i * 2 + 1] = rsqrtf(var + EPS);
}
// normalize + transpose (bhw,c) -> bf16 [b*64+c][hw]
__global__ __launch_bounds__(256) void k_in1_apply(const float* x, char* ws) {
  __shared__ __bf16 tile[256][66];
  int b = blockIdx.x >> 6, hw0 = (blockIdx.x & 63) * 256, t = threadIdx.x;
  const float* st = (const float*)(ws + OFF_ST1);
  for (int i = 0; i < 64; i++) {
    int idx = i * 256 + t;
    int hwl = idx >> 6, c = idx & 63;
    float m = st[(b * 64 + c) * 2], rs = st[(b * 64 + c) * 2 + 1];
    float v = x[((size_t)b * HW + hw0 + hwl) * 64 + c];
    tile[hwl][c] = (__bf16)((v - m) * rs);
  }
  __syncthreads();
  __bf16* xn = (__bf16*)(ws + OFF_XN);
  for (int c = 0; c < 64; c++) xn[((size_t)(b * 64 + c)) * HW + hw0 + t] = tile[t][c];
}

// ---------------- stage A: G = xn @ WAT ; [131072,128]@[128,32] -> GT[img][32][128] ----------------
__global__ __launch_bounds__(256) void k_stageA(char* ws) {
  int img = blockIdx.x;
  int wv = threadIdx.x >> 5, lane = threadIdx.x & 31;
  int l15 = lane & 15, hi8 = (lane & 16) ? 8 : 0;
  const __bf16* xn = (const __bf16*)(ws + OFF_XN) + (size_t)img * HW;
  const __bf16* wat = (const __bf16*)(ws + OFF_WAT);
  int h0 = wv * 16;
  v8f acc0 = {}, acc1 = {};
#pragma unroll
  for (int kk = 0; kk < 4; kk++) {
    v16bf a = load_a(xn + (size_t)h0 * 128 + kk * 32, 128);
    v16bf b0 = load_b(wat + kk * 32, 128);
    v16bf b1 = load_b(wat + 16 * 128 + kk * 32, 128);
    acc0 = wmma_bf16(a, b0, acc0);
    acc1 = wmma_bf16(a, b1, acc1);
  }
  __bf16* gt = (__bf16*)(ws + OFF_GT) + (size_t)img * 32 * 128;
  v8bf p0, p1;
#pragma unroll
  for (int r = 0; r < 8; r++) { p0[r] = (__bf16)acc0[r]; p1[r] = (__bf16)acc1[r]; }
  *(v8bf*)(gt + (size_t)l15 * 128 + h0 + hi8) = p0;
  *(v8bf*)(gt + (size_t)(l15 + 16) * 128 + h0 + hi8) = p1;
}

// ---------------- stage B: X[32ky][16kx] = FH(e^{-i}) @ G ; write Ar/Ai[mode][b][ci] ----------------
__global__ __launch_bounds__(256) void k_stageB(char* ws) {
  int img = blockIdx.x * 8 + (threadIdx.x >> 5);
  int lane = threadIdx.x & 31, l15 = lane & 15, hi8 = (lane & 16) ? 8 : 0;
  int b = img >> 6, ci = img & 63;
  const __bf16* gt = (const __bf16*)(ws + OFF_GT) + (size_t)img * 4096;
  const __bf16* fc = (const __bf16*)(ws + OFF_FBH);
  const __bf16* fs = fc + 4096;
  const __bf16* fn = fc + 8192;
  __bf16* Ar = (__bf16*)(ws + OFF_AR);
  __bf16* Ai = (__bf16*)(ws + OFF_AI);
#pragma unroll
  for (int mt = 0; mt < 2; mt++) {
    v8f xr = {}, xi = {};
#pragma unroll
    for (int kk = 0; kk < 4; kk++) {
      v16bf ac = load_a(fc + (size_t)mt * 16 * 128 + kk * 32, 128);
      v16bf as = load_a(fs + (size_t)mt * 16 * 128 + kk * 32, 128);
      v16bf an = load_a(fn + (size_t)mt * 16 * 128 + kk * 32, 128);
      v16bf br = load_b(gt + kk * 32, 128);            // Gr cols 0..15
      v16bf bi = load_b(gt + 2048 + kk * 32, 128);     // Gi cols 16..31
      xr = wmma_bf16(ac, br, xr);  // cos@Gr
      xr = wmma_bf16(as, bi, xr);  // + sin@Gi
      xi = wmma_bf16(ac, bi, xi);  // cos@Gi
      xi = wmma_bf16(an, br, xi);  // - sin@Gr
    }
    int kx = l15;
#pragma unroll
    for (int r = 0; r < 8; r++) {
      int j = mt * 16 + r + hi8;
      size_t o = ((size_t)(j * 16 + kx) * 16 + b) * 64 + ci;
      Ar[o] = (__bf16)xr[r];
      Ai[o] = (__bf16)xi[r];
    }
  }
}

// ---------------- stage C: per-mode complex GEMM [16b x 64ci] @ [64ci x 64co] ----------------
__global__ __launch_bounds__(256) void k_stageC(char* ws) {
  int mode = blockIdx.x * 8 + (threadIdx.x >> 5);
  int lane = threadIdx.x & 31, l15 = lane & 15, hi8 = (lane & 16) ? 8 : 0;
  const __bf16* ar = (const __bf16*)(ws + OFF_AR) + (size_t)mode * 1024;
  const __bf16* ai = (const __bf16*)(ws + OFF_AI) + (size_t)mode * 1024;
  v16bf ar0 = load_a(ar, 64), ar1 = load_a(ar + 32, 64);
  v16bf ai0 = load_a(ai, 64), ai1 = load_a(ai + 32, 64);
  int ky = mode >> 4, kx = mode & 15;
  __bf16* Ytr = (__bf16*)(ws + OFF_YTR);
  __bf16* Yti = (__bf16*)(ws + OFF_YTI);
#pragma unroll
  for (int n = 0; n < 4; n++) {
    const __bf16* wr = (const __bf16*)(ws + OFF_WR) + (size_t)mode * 4096 + n * 16 * 64;
    const __bf16* wi = (const __bf16*)(ws + OFF_WI) + (size_t)mode * 4096 + n * 16 * 64;
    const __bf16* wn = (const __bf16*)(ws + OFF_WN) + (size_t)mode * 4096 + n * 16 * 64;
    v8f yr = {}, yi = {};
    v16bf b0 = load_b(wr, 64), b1 = load_b(wr + 32, 64);
    yr = wmma_bf16(ar0, b0, yr); yr = wmma_bf16(ar1, b1, yr);   // Xr@Wr
    yi = wmma_bf16(ai0, b0, yi); yi = wmma_bf16(ai1, b1, yi);   // Xi@Wr
    b0 = load_b(wn, 64); b1 = load_b(wn + 32, 64);
    yr = wmma_bf16(ai0, b0, yr); yr = wmma_bf16(ai1, b1, yr);   // - Xi@Wi
    b0 = load_b(wi, 64); b1 = load_b(wi + 32, 64);
    yi = wmma_bf16(ar0, b0, yi); yi = wmma_bf16(ar1, b1, yi);   // + Xr@Wi
    int co = n * 16 + l15;
#pragma unroll
    for (int r = 0; r < 8; r++) {
      int bb = r + hi8;
      size_t o = (((size_t)(bb * 64 + co)) * 16 + kx) * 32 + ky;
      Ytr[o] = (__bf16)yr[r];
      Yti[o] = (__bf16)yi[r];
    }
  }
}

// ---------------- stage D: Z[h][kx] = FIH(e^{+i}) @ Y ; write Z[row][32] ----------------
__global__ __launch_bounds__(256) void k_stageD(char* ws) {
  int img2 = blockIdx.x * 8 + (threadIdx.x >> 5);
  int lane = threadIdx.x & 31, l15 = lane & 15, hi8 = (lane & 16) ? 8 : 0;
  const __bf16* fc = (const __bf16*)(ws + OFF_FIH);
  const __bf16* fs = fc + 4096;
  const __bf16* fn = fc + 8192;
  v16bf byr = load_b((const __bf16*)(ws + OFF_YTR) + (size_t)img2 * 512, 32);
  v16bf byi = load_b((const __bf16*)(ws + OFF_YTI) + (size_t)img2 * 512, 32);
  __bf16* Z = (__bf16*)(ws + OFF_Z);
#pragma unroll
  for (int mt = 0; mt < 8; mt++) {
    v16bf ac = load_a(fc + (size_t)mt * 16 * 32, 32);
    v16bf as = load_a(fs + (size_t)mt * 16 * 32, 32);
    v16bf an = load_a(fn + (size_t)mt * 16 * 32, 32);
    v8f zr = {}, zi = {};
    zr = wmma_bf16(ac, byr, zr);  // cos@Yr
    zr = wmma_bf16(an, byi, zr);  // - sin@Yi
    zi = wmma_bf16(ac, byi, zi);  // cos@Yi
    zi = wmma_bf16(as, byr, zi);  // + sin@Yr
#pragma unroll
    for (int r = 0; r < 8; r++) {
      int h = mt * 16 + r + hi8;
      size_t ro = ((size_t)img2 * 128 + h) * 32;
      Z[ro + l15] = (__bf16)zr[r];
      Z[ro + 16 + l15] = (__bf16)zi[r];
    }
  }
}

// ---------------- stage E: y = Z @ WET ; [131072,32]@[32,128] -> YS[img2][h][w] ----------------
__global__ __launch_bounds__(256) void k_stageE(char* ws) {
  int img2 = blockIdx.x;
  int mt = threadIdx.x >> 5, lane = threadIdx.x & 31;
  int l15 = lane & 15, hi8 = (lane & 16) ? 8 : 0;
  size_t row0 = (size_t)img2 * 128 + mt * 16;
  v16bf a = load_a((const __bf16*)(ws + OFF_Z) + row0 * 32, 32);
  const __bf16* wet = (const __bf16*)(ws + OFF_WET);
  __bf16* YS = (__bf16*)(ws + OFF_YS);
#pragma unroll
  for (int n = 0; n < 8; n++) {
    v16bf bw = load_b(wet + (size_t)n * 16 * 32, 32);
    v8f acc = {};
    acc = wmma_bf16(a, bw, acc);
    int w = n * 16 + l15;
#pragma unroll
    for (int r = 0; r < 8; r++) {
      int h = mt * 16 + r + hi8;
      YS[(size_t)img2 * HW + h * 128 + w] = (__bf16)acc[r];
    }
  }
}

// ---------------- instance norm 2 ----------------
__global__ __launch_bounds__(256) void k_in2_stats(char* ws) {
  __shared__ float ls[256], ls2[256];
  int img2 = blockIdx.x, t = threadIdx.x;
  const __bf16* YS = (const __bf16*)(ws + OFF_YS);
  float s = 0.f, s2 = 0.f;
  for (int i = 0; i < 64; i++) {
    float v = (float)YS[(size_t)img2 * HW + i * 256 + t];
    s += v; s2 += v * v;
  }
  ls[t] = s; ls2[t] = s2;
  __syncthreads();
  for (int off = 128; off >= 1; off >>= 1) {
    if (t < off) { ls[t] += ls[t + off]; ls2[t] += ls2[t + off]; }
    __syncthreads();
  }
  if (t == 0) {
    float m = ls[0] / 16384.0f;
    float var = ls2[0] / 16384.0f - m * m;
    float* st = (float*)(ws + OFF_ST2);
    st[img2 * 2] = m;
    st[img2 * 2 + 1] = rsqrtf(var + EPS);
  }
}
// normalize + transpose [b][c][hw] -> token-major bf16 [b*HW+hw][c]
__global__ __launch_bounds__(256) void k_in2_apply(char* ws) {
  __shared__ __bf16 tile[256][66];
  int b = blockIdx.x >> 6, hw0 = (blockIdx.x & 63) * 256, t = threadIdx.x;
  const __bf16* YS = (const __bf16*)(ws + OFF_YS);
  const float* st = (const float*)(ws + OFF_ST2);
  for (int c = 0; c < 64; c++) {
    float m = st[(b * 64 + c) * 2], rs = st[(b * 64 + c) * 2 + 1];
    float v = (float)YS[((size_t)(b * 64 + c)) * HW + hw0 + t];
    tile[t][c] = (__bf16)((v - m) * rs);
  }
  __syncthreads();
  __bf16* ym = (__bf16*)(ws + OFF_YMLP);
  for (int i = 0; i < 64; i++) {
    int idx = i * 256 + t;
    int hwl = idx >> 6, c = idx & 63;
    ym[((size_t)b * HW + hw0 + hwl) * 64 + c] = tile[hwl][c];
  }
}

// ---------------- MLP GEMM1: [262144,64]@[64,128] + bias + exact GELU ----------------
__global__ __launch_bounds__(256) void k_gemm1(char* ws, const float* mlp_b1) {
  int gid = blockIdx.x * 8 + (threadIdx.x >> 5);
  int lane = threadIdx.x & 31, l15 = lane & 15, hi8 = (lane & 16) ? 8 : 0;
  size_t t0 = (size_t)gid * 16;
  const __bf16* A = (const __bf16*)(ws + OFF_YMLP) + t0 * 64;
  const __bf16* W1 = (const __bf16*)(ws + OFF_W1B);
  __bf16* HACT = (__bf16*)(ws + OFF_HACT);
  v16bf a0 = load_a(A, 64), a1 = load_a(A + 32, 64);
#pragma unroll
  for (int n = 0; n < 8; n++) {
    v8f acc = {};
    acc = wmma_bf16(a0, load_b(W1 + (size_t)n * 16 * 64, 64), acc);
    acc = wmma_bf16(a1, load_b(W1 + (size_t)n * 16 * 64 + 32, 64), acc);
    int o = n * 16 + l15;
    float bias = mlp_b1[o];
#pragma unroll
    for (int r = 0; r < 8; r++) {
      float v = acc[r] + bias;
      v = 0.5f * v * (1.0f + erff(v * 0.70710678118654752f));
      HACT[(t0 + r + hi8) * 128 + o] = (__bf16)v;
    }
  }
}

// ---------------- MLP GEMM2: [262144,128]@[128,64] + bias -> f32 out ----------------
__global__ __launch_bounds__(256) void k_gemm2(char* ws, const float* mlp_b2, float* out) {
  int gid = blockIdx.x * 8 + (threadIdx.x >> 5);
  int lane = threadIdx.x & 31, l15 = lane & 15, hi8 = (lane & 16) ? 8 : 0;
  size_t t0 = (size_t)gid * 16;
  const __bf16* A = (const __bf16*)(ws + OFF_HACT) + t0 * 128;
  const __bf16* W2 = (const __bf16*)(ws + OFF_W2B);
  v16bf a0 = load_a(A, 128), a1 = load_a(A + 32, 128);
  v16bf a2 = load_a(A + 64, 128), a3 = load_a(A + 96, 128);
#pragma unroll
  for (int n = 0; n < 4; n++) {
    const __bf16* wb = W2 + (size_t)n * 16 * 128;
    v8f acc = {};
    acc = wmma_bf16(a0, load_b(wb, 128), acc);
    acc = wmma_bf16(a1, load_b(wb + 32, 128), acc);
    acc = wmma_bf16(a2, load_b(wb + 64, 128), acc);
    acc = wmma_bf16(a3, load_b(wb + 96, 128), acc);
    int c = n * 16 + l15;
    float bias = mlp_b2[c];
#pragma unroll
    for (int r = 0; r < 8; r++) out[(t0 + r + hi8) * 64 + c] = acc[r] + bias;
  }
}

// ---------------- host launcher ----------------
extern "C" void kernel_launch(void* const* d_in, const int* in_sizes, int n_in,
                              void* d_out, int out_size, void* d_ws, size_t ws_size,
                              hipStream_t stream) {
  const float* x = (const float*)d_in[0];
  const float* w1r = (const float*)d_in[1];
  const float* w1i = (const float*)d_in[2];
  const float* w2r = (const float*)d_in[3];
  const float* w2i = (const float*)d_in[4];
  const float* mlp_w1 = (const float*)d_in[5];
  const float* mlp_b1 = (const float*)d_in[6];
  const float* mlp_w2 = (const float*)d_in[7];
  const float* mlp_b2 = (const float*)d_in[8];
  char* ws = (char*)d_ws;
  float* out = (float*)d_out;

  k_init_tables<<<dim3(192), dim3(256), 0, stream>>>(ws, mlp_w1, mlp_w2);
  k_packw<<<dim3(8192), dim3(256), 0, stream>>>(ws, w1r, w1i, w2r, w2i);
  k_in1_partial<<<dim3(1024), dim3(256), 0, stream>>>(x, ws);
  k_in1_final<<<dim3(4), dim3(256), 0, stream>>>(ws);
  k_in1_apply<<<dim3(1024), dim3(256), 0, stream>>>(x, ws);
  k_stageA<<<dim3(1024), dim3(256), 0, stream>>>(ws);
  k_stageB<<<dim3(128), dim3(256), 0, stream>>>(ws);
  k_stageC<<<dim3(64), dim3(256), 0, stream>>>(ws);
  k_stageD<<<dim3(128), dim3(256), 0, stream>>>(ws);
  k_stageE<<<dim3(1024), dim3(256), 0, stream>>>(ws);
  k_in2_stats<<<dim3(1024), dim3(256), 0, stream>>>(ws);
  k_in2_apply<<<dim3(1024), dim3(256), 0, stream>>>(ws);
  k_gemm1<<<dim3(2048), dim3(256), 0, stream>>>(ws, mlp_b1);
  k_gemm2<<<dim3(2048), dim3(256), 0, stream>>>(ws, mlp_b2, out);
}
